// FFDNN_33363305955785
// MI455X (gfx1250) — matrix operand order repaired and checked by reading, not compiled, not run on hardware
//
#include <hip/hip_runtime.h>
#include <stdint.h>

// ---------------------------------------------------------------------------
// CDNA5 / gfx1250 (MI455X): bf16 WMMA GEMM chain for the quadrature FFDNN.
// - Activations flow between layers as bf16 with trapezoid weights pre-folded
//   (previous layer's epilogue produces them); main loop converts only weight
//   tiles via native v_cvt_pk_bf16_f32.
// - A-tile staging via Tensor Data Mover (tensor_load_to_lds), manual fallback.
// - Double-buffered LDS, one barrier per K-chunk, incremental W addressing.
// - Layer 3 (tiny N) uses split-K across blockIdx.z + deterministic reduce.
// ---------------------------------------------------------------------------

typedef __attribute__((ext_vector_type(16))) __bf16 v16bf;
typedef __attribute__((ext_vector_type(8)))  float  v8f;

#if defined(__has_builtin)
#  if __has_builtin(__builtin_amdgcn_tensor_load_to_lds)
#    define HAS_TDM 1
#  endif
#  if __has_builtin(__builtin_amdgcn_cvt_pk_bf16_f32)
#    define HAS_PKBF 1
#  endif
#endif
#ifndef HAS_TDM
#  define HAS_TDM 0
#endif
#ifndef HAS_PKBF
#  define HAS_PKBF 0
#endif

union Frag16 { v16bf v; uint2 u2[4]; };     // 16 bf16 = 8 VGPRs
union Acc8   { v8f   v; float f[8]; };

#define STRD 40   // LDS row stride in bf16 elems: 32 payload + 8 pad (80B = 16B-aligned, conflict-free)

// pack two f32 into two bf16 (lo|hi<<16)
__device__ __forceinline__ unsigned pk2bf(float lo, float hi) {
#if HAS_PKBF
    auto r = __builtin_amdgcn_cvt_pk_bf16_f32(lo, hi);   // v_cvt_pk_bf16_f32
    unsigned u;
    __builtin_memcpy(&u, &r, 4);
    return u;
#else
    union { __bf16 b[2]; unsigned u; } c;
    c.b[0] = (__bf16)lo;
    c.b[1] = (__bf16)hi;
    return c.u;
#endif
}

#if HAS_TDM
typedef __attribute__((ext_vector_type(4))) unsigned int tdm_v4u;
typedef __attribute__((ext_vector_type(8))) int          tdm_v8i;
typedef __attribute__((ext_vector_type(4))) int          tdm_v4i;

// DMA a 128-row x 32-col bf16 tile (row stride Kdim elems) into LDS at
// lds_off, inserting 4 DWORDs of pad after every 16 DWORDs -> 80B LDS rows.
__device__ __forceinline__ void tdm_load_tile(const unsigned short* gsrc,
                                              unsigned lds_off, int Kdim) {
    unsigned long long ga = (unsigned long long)(uintptr_t)gsrc;
    tdm_v4u g0;
    g0[0] = 1u;                                   // count=1 (valid descriptor)
    g0[1] = lds_off;                              // lds_addr (bytes)
    g0[2] = (unsigned)ga;                         // global_addr[31:0]
    g0[3] = ((unsigned)(ga >> 32) & 0x01FFFFFFu)  // global_addr[56:32]
            | 0x80000000u;                        // type=2 ("image")
    tdm_v8i g1;
    g1[0] = (int)((1u << 16)                      // data_size = 2 bytes
                | (1u << 20)                      // pad_enable
                | (3u << 22)                      // pad_interval: 16 DWORDs
                | (3u << 25));                    // pad_amount: 4 DWORDs
    g1[1] = (int)(((unsigned)Kdim & 0xFFFFu) << 16);            // tensor_dim0 lo16
    g1[2] = (int)((((unsigned)Kdim >> 16) & 0xFFFFu));          // tensor_dim0 hi | dim1 lo16 (=0)
    g1[3] = (int)(1u | (32u << 16));              // tensor_dim1 hi16 (=65536) | tile_dim0=32
    g1[4] = 128;                                  // tile_dim1 = 128 rows
    g1[5] = Kdim;                                 // tensor_dim0_stride lo32
    g1[6] = 0;
    g1[7] = 0;
    tdm_v4i gz4 = {0, 0, 0, 0};
    tdm_v8i gz8 = {0, 0, 0, 0, 0, 0, 0, 0};
    __builtin_amdgcn_tensor_load_to_lds(g0, g1, gz4, gz4, gz8, 0);
}
#endif

// ---- W tile staging: 32 k-rows x 128 n-cols, incremental element offsets --
__device__ __forceinline__ void load_w_regs(const float* __restrict__ W,
                                            float4 (&wreg)[4],
                                            const unsigned (&woff)[4]) {
#pragma unroll
    for (int pass = 0; pass < 4; ++pass)
        wreg[pass] = *(const float4*)(W + woff[pass]);
}

__device__ __forceinline__ void store_w_lds(unsigned short* __restrict__ dst,
                                            const float4 (&wreg)[4], int t) {
#pragma unroll
    for (int pass = 0; pass < 4; ++pass) {
        int idx = pass * 256 + t;
        int kl  = idx >> 5;                  // 0..31
        int nl  = (idx & 31) << 2;           // 0..124
        const float* w = (const float*)&wreg[pass];
        unsigned u01 = pk2bf(w[0], w[1]);
        unsigned u23 = pk2bf(w[2], w[3]);
        dst[(nl + 0) * STRD + kl] = (unsigned short)u01;           // ds_store_b16
        dst[(nl + 1) * STRD + kl] = (unsigned short)(u01 >> 16);   // ds_store_b16_d16_hi
        dst[(nl + 2) * STRD + kl] = (unsigned short)u23;
        dst[(nl + 3) * STRD + kl] = (unsigned short)(u23 >> 16);
    }
}

#if !HAS_TDM
// ---- manual A tile staging (bf16 copy, no conversion) ---------------------
__device__ __forceinline__ void load_a_regs(const unsigned short* __restrict__ A,
                                            uint4 (&areg)[2], int t, int m0,
                                            int kk0, int Kdim) {
#pragma unroll
    for (int pass = 0; pass < 2; ++pass) {
        int idx = pass * 256 + t;            // 0..511
        int r   = idx >> 2;                  // 0..127
        int c   = idx & 3;                   // 4 x 8 bf16 per row
        areg[pass] = *(const uint4*)(A + (size_t)(m0 + r) * Kdim + kk0 + c * 8);
    }
}
__device__ __forceinline__ void store_a_lds(unsigned short* __restrict__ dst,
                                            const uint4 (&areg)[2], int t) {
#pragma unroll
    for (int pass = 0; pass < 2; ++pass) {
        int idx = pass * 256 + t;
        int r   = idx >> 2;
        int c   = idx & 3;
        *(uint4*)(dst + r * STRD + c * 8) = areg[pass];   // 80r+16c bytes: 16B aligned
    }
}
#endif

// H[m,n] = bias[n] + sum_kk A_bf[m,kk] * W[((j*Kout+ko)*P+p)*Q+q]
//   kk = j*P+p, n = ko*Q+q.  A_bf already carries the trapezoid weights.
// OUTBF=1: out = bf16( relu(H) * tw_next(n) )         (feeds next layer)
// OUTBF=0: f32 partial sums per K-slice (blockIdx.z); bias added in reduce.
template <int PSH, int QSH, int OUTBF>
__global__ __launch_bounds__(256)
void ffidnn_gemm(const unsigned short* __restrict__ A, const float* __restrict__ W,
                 const float* __restrict__ bias, void* __restrict__ OutP,
                 int N, int Kdim, int Kout, float dxo, int kslice)
{
    constexpr int P = 1 << PSH, Q = 1 << QSH;
    __shared__ unsigned short Al[2][128 * STRD];
    __shared__ unsigned short Wl[2][128 * STRD];

    const int t = threadIdx.x, lane = t & 31, wave = t >> 5;
    const int wm = wave & 3;       // wave's 32-row group
    const int wn = wave >> 2;      // wave's 64-col group
    const int m0 = blockIdx.x * 128, n0 = blockIdx.y * 128;
    const int kstart = blockIdx.z * kslice;   // K-slice origin (elements)
    const int kcount = kslice >> 5;           // chunks in this slice

    Acc8 acc[2][4];
#pragma unroll
    for (int im = 0; im < 2; ++im)
#pragma unroll
        for (int in = 0; in < 4; ++in)
#pragma unroll
            for (int r = 0; r < 8; ++r) acc[im][in].f[r] = 0.0f;

    // per-pass W element offsets for the slice's first chunk
    // (kstart is a multiple of 32, so p0 = kstart & (P-1) keeps p0+kl < P)
    unsigned woff[4];
    {
        const unsigned j0 = (unsigned)kstart >> PSH;
        const unsigned p0 = (unsigned)kstart & (unsigned)(P - 1);
#pragma unroll
        for (int pass = 0; pass < 4; ++pass) {
            int idx = pass * 256 + t;
            int kl  = idx >> 5;
            int nl  = (idx & 31) << 2;
            int n   = n0 + nl;
            int ko  = n >> QSH;
            int q   = n & (Q - 1);
            woff[pass] = j0 * ((unsigned)Kout << (PSH + QSH))
                       + ((unsigned)ko << (PSH + QSH))
                       + ((p0 + (unsigned)kl) << QSH) + (unsigned)q;
        }
    }
    const unsigned stepN    = 32u << QSH;                                    // same j
    const unsigned stepWrap = ((unsigned)(Kout - 1) << (PSH + QSH)) + stepN; // j+1, p-=P

    float4 wreg[4];
#if !HAS_TDM
    uint4 areg[2];
#endif

    // ---- prologue: stage chunk 0 into buffer 0 ----
#if HAS_TDM
    if (wave == 0)
        tdm_load_tile(A + (size_t)m0 * Kdim + kstart,
                      (unsigned)(uintptr_t)&Al[0][0], Kdim);
#else
    load_a_regs(A, areg, t, m0, kstart, Kdim);
#endif
    load_w_regs(W, wreg, woff);
    {   // advance offsets to chunk 1
        unsigned step =
            ((((unsigned)(kstart + 32)) & (unsigned)(P - 1)) == 0u) ? stepWrap : stepN;
#pragma unroll
        for (int pass = 0; pass < 4; ++pass) woff[pass] += step;
    }
#if !HAS_TDM
    store_a_lds(&Al[0][0], areg, t);
#endif
    store_w_lds(&Wl[0][0], wreg, t);
#if HAS_TDM
    if (wave == 0) __builtin_amdgcn_s_wait_tensorcnt(0);
#endif
    __syncthreads();

    for (int ch = 0; ch < kcount; ++ch) {
        const int  cur  = ch & 1;
        const int  nxtb = cur ^ 1;
        const bool more = (ch + 1) < kcount;

        // issue next chunk's loads first (latency hidden by WMMAs below)
        if (more) {
#if HAS_TDM
            if (wave == 0)
                tdm_load_tile(A + (size_t)m0 * Kdim + kstart + (ch + 1) * 32,
                              (unsigned)(uintptr_t)&Al[nxtb][0], Kdim);
#else
            load_a_regs(A, areg, t, m0, kstart + (ch + 1) * 32, Kdim);
#endif
            load_w_regs(W, wreg, woff);
            unsigned step =
                ((((unsigned)(kstart + (ch + 2) * 32)) & (unsigned)(P - 1)) == 0u)
                    ? stepWrap : stepN;
#pragma unroll
            for (int pass = 0; pass < 4; ++pass) woff[pass] += step;
        }

        // ---- compute: 8 WMMAs (2 M-subtiles x 4 N-subtiles) per wave ----
        {
            const unsigned short* Ab = &Al[cur][0];
            const unsigned short* Wb = &Wl[cur][0];
            const int h8  = (lane >> 4) << 3;   // A frag half select
            const int h16 = (lane >> 4) << 4;   // unused; B frags also use h8
            Frag16 a[2];
#pragma unroll
            for (int im = 0; im < 2; ++im) {
                const unsigned short* ab = Ab + (wm * 32 + im * 16 + (lane & 15)) * STRD;
                a[im].u2[0] = *(const uint2*)(ab + h8);        // VGPR0-1
                a[im].u2[1] = *(const uint2*)(ab + h8 + 4);    // VGPR2-3
                a[im].u2[2] = *(const uint2*)(ab + 16 + h8);   // VGPR4-5
                a[im].u2[3] = *(const uint2*)(ab + 20 + h8);   // VGPR6-7
            }
#pragma unroll
            for (int in = 0; in < 4; ++in) {
                Frag16 b;
                const unsigned short* bb =
                    Wb + (wn * 64 + in * 16 + (lane & 15)) * STRD + h8;
                b.u2[0] = *(const uint2*)(bb);
                b.u2[1] = *(const uint2*)(bb + 4);
                b.u2[2] = *(const uint2*)(bb + 16);
                b.u2[3] = *(const uint2*)(bb + 20);
                acc[0][in].v = __builtin_amdgcn_wmma_f32_16x16x32_bf16(
                    false, a[0].v, false, b.v, (short)0, acc[0][in].v, false, false);
                acc[1][in].v = __builtin_amdgcn_wmma_f32_16x16x32_bf16(
                    false, a[1].v, false, b.v, (short)0, acc[1][in].v, false, false);
            }
        }

        // commit next chunk's tiles
        if (more) {
#if !HAS_TDM
            store_a_lds(&Al[nxtb][0], areg, t);
#endif
            store_w_lds(&Wl[nxtb][0], wreg, t);
#if HAS_TDM
            if (wave == 0) __builtin_amdgcn_s_wait_tensorcnt(0);
#endif
        }
        __syncthreads();
    }

    // ---- epilogue ----
    const int   rowb    = m0 + wm * 32 + ((lane >> 4) << 3);
    const int   colb    = n0 + wn * 64 + (lane & 15);
    const float halfdxo = 0.5f * dxo;
#pragma unroll
    for (int in = 0; in < 4; ++in) {
        const int col = colb + (in << 4);
#pragma unroll
        for (int im = 0; im < 2; ++im) {
            const int row0 = rowb + im * 16;
            if (OUTBF) {
                const float bv = bias[col];
                int   p = col & 127;                      // next-layer grid pos
                float w = (p == 0 || p == 127) ? halfdxo : dxo;
#pragma unroll
                for (int r = 0; r < 8; r += 2) {
                    float v0 = fmaxf(acc[im][in].f[r] + bv, 0.0f) * w;
                    float v1 = fmaxf(acc[im][in].f[r + 1] + bv, 0.0f) * w;
                    unsigned u = pk2bf(v0, v1);
                    unsigned short* o = (unsigned short*)OutP;
                    o[(size_t)(row0 + r) * N + col]     = (unsigned short)u;
                    o[(size_t)(row0 + r + 1) * N + col] = (unsigned short)(u >> 16);
                }
            } else {
                // f32 partial for this K-slice; bias added during reduction
                float* po = (float*)OutP + (size_t)blockIdx.z * ((size_t)256 * N);
#pragma unroll
                for (int r = 0; r < 8; ++r)
                    po[(size_t)(row0 + r) * N + col] = acc[im][in].f[r];
            }
        }
    }
}

// Layer-0 activation prep: fold trapezoid weights (P=64 grid), cvt to bf16.
// 2 elements per thread -> packed b32 stores.
__global__ __launch_bounds__(256)
void prep_a0(const float* __restrict__ x, unsigned short* __restrict__ a0,
             int total2, float dx) {
    int i2 = blockIdx.x * blockDim.x + threadIdx.x;
    if (i2 >= total2) return;
    int   i  = i2 * 2;
    int   p0 = i & 63, p1 = (i + 1) & 63;
    float w0 = (p0 == 0 || p0 == 63) ? 0.5f * dx : dx;
    float w1 = (p1 == 0 || p1 == 63) ? 0.5f * dx : dx;
    *(unsigned*)&a0[i] = pk2bf(x[i] * w0, x[i + 1] * w1);
}

// Deterministic split-K reduction: out[i] = bias[i % N] + sum_z part[z][i]
__global__ __launch_bounds__(256)
void reduce_out(const float* __restrict__ part, const float* __restrict__ bias,
                float* __restrict__ out, int total, int N, int slices) {
    int i = blockIdx.x * blockDim.x + threadIdx.x;
    if (i >= total) return;
    float s = bias[i & (N - 1)];
    for (int z = 0; z < slices; ++z)
        s += part[(size_t)z * total + i];
    out[i] = s;
}

extern "C" void kernel_launch(void* const* d_in, const int* in_sizes, int n_in,
                              void* d_out, int out_size, void* d_ws, size_t ws_size,
                              hipStream_t stream) {
    const float* x  = (const float*)d_in[0];
    const float* W0 = (const float*)d_in[1];
    const float* B0 = (const float*)d_in[2];
    const float* W1 = (const float*)d_in[3];
    const float* B1 = (const float*)d_in[4];
    const float* W2 = (const float*)d_in[5];
    const float* B2 = (const float*)d_in[6];
    const float* W3 = (const float*)d_in[7];
    const float* B3 = (const float*)d_in[8];
    float* out = (float*)d_out;

    unsigned short* a0   = (unsigned short*)d_ws;         // 256x256  bf16 (128 KB)
    unsigned short* hb1  = a0 + (size_t)256 * 256;        // 256x8192 bf16 (4 MB)
    unsigned short* hb2  = hb1 + (size_t)256 * 8192;      // 256x8192 bf16 (4 MB)
    float*          part = (float*)(hb2 + (size_t)256 * 8192); // 32 x 65536 f32 (8 MB)

    const float DXI = 1.0f / 63.0f;   // 1/(Q_IN-1)
    const float DXH = 1.0f / 127.0f;  // 1/(Q_HID-1)
    const int   SLICES = 32;          // layer-3 split-K factor (8192/32 = 256 elems)

    dim3 blk(256);
    prep_a0<<<dim3(128), blk, 0, stream>>>(x, a0, 256 * 256 / 2, DXI);
    // L0: K=256 (P=64, PSH=6), N=8192 (Q=128), Kout=64 -> bf16+tw out
    ffidnn_gemm<6, 7, 1><<<dim3(2, 64), blk, 0, stream>>>(a0,  W0, B0, hb1, 8192, 256,  64, DXH, 256);
    // L1, L2: K=8192 (P=128), N=8192
    ffidnn_gemm<7, 7, 1><<<dim3(2, 64), blk, 0, stream>>>(hb1, W1, B1, hb2, 8192, 8192, 64, DXH, 8192);
    ffidnn_gemm<7, 7, 1><<<dim3(2, 64), blk, 0, stream>>>(hb2, W2, B2, hb1, 8192, 8192, 64, DXH, 8192);
    // L3: N=256 (Q=64), Kout=4 -> split-K f32 partials, then reduce (+bias)
    ffidnn_gemm<7, 6, 0><<<dim3(2, 2, SLICES), blk, 0, stream>>>(hb1, W3, B3, part, 256, 8192, 4, DXH, 8192 / SLICES);
    reduce_out<<<dim3(256), blk, 0, stream>>>(part, B3, out, 256 * 256, 256, SLICES);
}
